// ROIHead_23854248362772
// MI455X (gfx1250) — matrix operand order repaired and checked by reading, not compiled, not run on hardware
//
#include <hip/hip_runtime.h>
#include <hip/hip_bf16.h>
#include <math.h>

// ---------------------------------------------------------------------------
// Types / helpers
// ---------------------------------------------------------------------------
typedef __bf16 v16bf __attribute__((ext_vector_type(16)));
typedef float  v8f   __attribute__((ext_vector_type(8)));

__device__ __forceinline__ unsigned int rnd_bf16(float f) {   // RNE, result in [31:16]
    unsigned int u = __float_as_uint(f);
    return u + 0x7FFFu + ((u >> 16) & 1u);
}
// pack two floats -> two bf16 in one dword (a -> low, b -> high), pure 32-bit VALU
__device__ __forceinline__ unsigned int pack2_bf16(float a, float b) {
    return (rnd_bf16(a) >> 16) | (rnd_bf16(b) & 0xFFFF0000u);
}
__device__ __forceinline__ unsigned short f2bf(float f) {
    return (unsigned short)(rnd_bf16(f) >> 16);
}
__device__ __forceinline__ float bf2f(unsigned short h) {
    return __uint_as_float(((unsigned int)h) << 16);
}

// ---------------------------------------------------------------------------
// 0) feat (B,256,32,32) -> featT (B,32,32,256)  so ROI-pool reads coalesce in C
// ---------------------------------------------------------------------------
__global__ void transpose_feat_kernel(const float* __restrict__ feat,
                                      float* __restrict__ featT, int B) {
    int i = blockIdx.x * blockDim.x + threadIdx.x;
    int total = B * 256 * 1024;
    if (i >= total) return;
    int c = i & 255;
    int rest = i >> 8;            // b*1024 + y*32 + x
    int yx = rest & 1023;
    int b  = rest >> 10;
    featT[i] = feat[((size_t)b * 256 + c) * 1024 + yx];
}

// ---------------------------------------------------------------------------
// 1) ROI max-pool (legacy RoIPool) -> X (B*R, 256*49) bf16, coalesced via LDS
// ---------------------------------------------------------------------------
__global__ __launch_bounds__(256)
void roi_pool_kernel(const float* __restrict__ featT,   // B,32,32,256
                     const float* __restrict__ props,   // B,R,4
                     unsigned short* __restrict__ X,    // (B*R) x 12544 bf16
                     int R) {
    const int C = 256, H = 32, W = 32;
    __shared__ __align__(16) unsigned short rowbuf[256 * 49];
    int br = blockIdx.x;
    int b  = br / R;
    const float* pr = props + (size_t)br * 4;
    const float scale = 1.0f / 16.0f;
    int x1 = (int)rintf(pr[0] * scale);
    int y1 = (int)rintf(pr[1] * scale);
    int x2 = (int)rintf(pr[2] * scale);
    int y2 = (int)rintf(pr[3] * scale);
    float roi_w = (float)max(x2 - x1 + 1, 1);
    float roi_h = (float)max(y2 - y1 + 1, 1);
    float bh = roi_h / 7.0f, bw = roi_w / 7.0f;
    int hs[7], he[7], ws[7], we[7];
#pragma unroll
    for (int p = 0; p < 7; ++p) {
        hs[p] = min(max((int)floorf((float)p * bh) + y1, 0), H);
        he[p] = min(max((int)ceilf((float)(p + 1) * bh) + y1, 0), H);
        ws[p] = min(max((int)floorf((float)p * bw) + x1, 0), W);
        we[p] = min(max((int)ceilf((float)(p + 1) * bw) + x1, 0), W);
    }
    int c = threadIdx.x;               // one channel per thread
    const float* fb = featT + (size_t)b * H * W * C;
    for (int py = 0; py < 7; ++py) {
        int h0 = hs[py], h1 = min(he[py], hs[py] + 8);   // MAXB = 8 cap
        for (int px = 0; px < 7; ++px) {
            int w0 = ws[px], w1 = min(we[px], ws[px] + 8);
            bool empty = (he[py] <= hs[py]) || (we[px] <= ws[px]);
            float m = -1e30f;
            for (int h = h0; h < h1; ++h)
                for (int w = w0; w < w1; ++w)
                    m = fmaxf(m, fb[((size_t)h * W + w) * C + c]);
            rowbuf[c * 49 + py * 7 + px] = f2bf(empty ? 0.0f : m);
        }
    }
    __syncthreads();
    // coalesced row store
    const unsigned int* src = (const unsigned int*)rowbuf;
    unsigned int* dst = (unsigned int*)(X + (size_t)br * (256 * 49));
    for (int i = threadIdx.x; i < (256 * 49) / 2; i += 256) dst[i] = src[i];
}

// ---------------------------------------------------------------------------
// 2) NT GEMM: C(MxN) = relu?(A(MxK,bf16) * B(NxK,f32->bf16)^T + bias)
//    Tiles 64x64x32, 4 waves, 2x2 wmma tiles/wave, double-buffered LDS,
//    one barrier per K-iteration, loads issued in the shadow of WMMA.
// ---------------------------------------------------------------------------
#define BM 64
#define BN 64
#define BK 32
#define LDT 40                // padded LDS row stride in ushorts (80B)
#define TSZ (BM * LDT)        // one tile buffer, in ushorts

__global__ __launch_bounds__(128)
void gemm_bf16_nt_kernel(const unsigned short* __restrict__ A,  // M x K bf16
                         const float* __restrict__ Bw,          // N x K f32
                         const float* __restrict__ bias,        // N
                         unsigned short* __restrict__ C,        // M x N bf16
                         int M, int N, int K, int relu) {
    __shared__ __align__(16) unsigned short As[2 * TSZ];
    __shared__ __align__(16) unsigned short Bs[2 * TSZ];
    const int tid  = threadIdx.x;
    const int lane = tid & 31;
    const int wave = tid >> 5;
    const int wm   = (wave >> 1) * 32;   // wave row offset in tile
    const int wn   = (wave & 1)  * 32;   // wave col offset in tile
    const int m0   = blockIdx.y * BM;
    const int n0   = blockIdx.x * BN;

    // ---- per-thread staging geometry (fixed for whole kernel) ----
    // A: 64 rows x 32 cols bf16 -> 256 16B-chunks; 128 thr x 2
    const int ar0 = tid >> 2,          ac0 = (tid & 3) * 8;
    const int ar1 = (tid + 128) >> 2,  ac1 = ((tid + 128) & 3) * 8;
    const unsigned short* ag0 = A + (size_t)(m0 + ar0) * K + ac0;
    const unsigned short* ag1 = A + (size_t)(m0 + ar1) * K + ac1;
    const int aso0 = ar0 * LDT + ac0;
    const int aso1 = ar1 * LDT + ac1;
    // B: 64 rows x 32 cols f32 -> 512 float4-chunks; 128 thr x 4
    const float* bg[4];
    int bso[4];
#pragma unroll
    for (int p = 0; p < 4; ++p) {
        int chunk = tid + p * 128;
        int r = chunk >> 2, c4 = (chunk & 3) * 4;
        bg[p]  = Bw + (size_t)(n0 + r) * K + c4;
        bso[p] = r * LDT + c4;
    }
    // fragment geometry
    const int mrow = lane & 15;
    const int kba  = (lane < 16) ? 0 : 16;   // A-frag K base (contiguous, matches B)
    const int kbb  = (lane < 16) ? 0 : 16;   // B-frag K base
    const int frA0 = (wm + 0 * 16 + mrow) * LDT;
    const int frA1 = (wm + 1 * 16 + mrow) * LDT;
    const int frB0 = (wn + 0 * 16 + mrow) * LDT;
    const int frB1 = (wn + 1 * 16 + mrow) * LDT;

    v8f acc[2][2];
    acc[0][0] = {}; acc[0][1] = {}; acc[1][0] = {}; acc[1][1] = {};

    // ---- prologue: load tile 0 into registers ----
    uint4 a0 = *(const uint4*)(ag0);
    uint4 a1 = *(const uint4*)(ag1);
    uint2 pb[4];
#pragma unroll
    for (int p = 0; p < 4; ++p) {
        float4 v = *(const float4*)(bg[p]);
        pb[p].x = pack2_bf16(v.x, v.y);
        pb[p].y = pack2_bf16(v.z, v.w);
    }

    const int nT = K / BK;
    for (int it = 0; it < nT; ++it) {
        unsigned short* as = As + (it & 1) * TSZ;
        unsigned short* bs = Bs + (it & 1) * TSZ;
        // 1) commit staged registers to LDS (loadcnt wait lands here)
        *(uint4*)(as + aso0) = a0;
        *(uint4*)(as + aso1) = a1;
#pragma unroll
        for (int p = 0; p < 4; ++p) *(uint2*)(bs + bso[p]) = pb[p];
        __syncthreads();
        // 2) issue next tile's global loads; no wait until next iteration
        if (it + 1 < nT) {
            const int ko = (it + 1) * BK;
            a0 = *(const uint4*)(ag0 + ko);
            a1 = *(const uint4*)(ag1 + ko);
#pragma unroll
            for (int p = 0; p < 4; ++p) {
                float4 v = *(const float4*)(bg[p] + ko);
                pb[p].x = pack2_bf16(v.x, v.y);
                pb[p].y = pack2_bf16(v.z, v.w);
            }
            if (it + 2 < nT) {   // prefetch tile it+2 (global_prefetch_b8)
                __builtin_prefetch(ag0 + ko + BK, 0, 1);
                __builtin_prefetch(bg[tid & 3] + ko + BK, 0, 1);
            }
        }
        // 3) fragments + WMMA from LDS[buf]
        union { v16bf v; unsigned short u[16]; } af[2], bfr[2];
        {
            const unsigned short* r0 = as + frA0;
            const unsigned short* r1 = as + frA1;
#pragma unroll
            for (int i = 0; i < 8; ++i) {
                af[0].u[i]     = r0[kba + i];
                af[0].u[i + 8] = r0[kba + 8 + i];
                af[1].u[i]     = r1[kba + i];
                af[1].u[i + 8] = r1[kba + 8 + i];
            }
            const unsigned short* r2 = bs + frB0;
            const unsigned short* r3 = bs + frB1;
#pragma unroll
            for (int i = 0; i < 16; ++i) {
                bfr[0].u[i] = r2[kbb + i];
                bfr[1].u[i] = r3[kbb + i];
            }
        }
#pragma unroll
        for (int tm = 0; tm < 2; ++tm)
#pragma unroll
            for (int tn = 0; tn < 2; ++tn)
                acc[tm][tn] = __builtin_amdgcn_wmma_f32_16x16x32_bf16(
                    false, af[tm].v, false, bfr[tn].v,
                    (short)0, acc[tm][tn], false, false);
    }

    // epilogue: D layout — VGPR v: M = v + 8*(lane>=16), N = lane&15
    const int mhalf = (lane >> 4) * 8;
    const int nn    = lane & 15;
#pragma unroll
    for (int tm = 0; tm < 2; ++tm)
#pragma unroll
        for (int tn = 0; tn < 2; ++tn) {
            int n = n0 + wn + tn * 16 + nn;
            float bv = bias[n];
#pragma unroll
            for (int v = 0; v < 8; ++v) {
                int m = m0 + wm + tm * 16 + v + mhalf;
                float x = acc[tm][tn][v] + bv;
                if (relu) x = fmaxf(x, 0.0f);
                C[(size_t)m * N + n] = f2bf(x);
            }
        }
}

// ---------------------------------------------------------------------------
// 3) cls/reg heads: logits (M,21), reg (M,84); one thread per (row, output)
// ---------------------------------------------------------------------------
__global__ void head_kernel(const unsigned short* __restrict__ Y2,
                            const float* __restrict__ Wc, const float* __restrict__ bc,
                            const float* __restrict__ Wb, const float* __restrict__ bb,
                            float* __restrict__ logits, float* __restrict__ reg,
                            int Mtot) {
    int idx = blockIdx.x * blockDim.x + threadIdx.x;
    if (idx >= Mtot * 105) return;
    int m = idx / 105, j = idx % 105;
    const unsigned short* yr = Y2 + (size_t)m * 1024;
    const float* w;
    float s;
    if (j < 21) { w = Wc + (size_t)j * 1024; s = bc[j]; }
    else        { w = Wb + (size_t)(j - 21) * 1024; s = bb[j - 21]; }
    for (int k = 0; k < 1024; ++k) s += bf2f(yr[k]) * w[k];
    if (j < 21) logits[(size_t)m * 21 + j] = s;
    else        reg[(size_t)m * 84 + (j - 21)] = s;
}

// ---------------------------------------------------------------------------
// 4) softmax + box decode + clip -> fg scores (B,20,R), fg boxes (B,20,R,4)
// ---------------------------------------------------------------------------
__global__ __launch_bounds__(32)
void decode_kernel(const float* __restrict__ logits, const float* __restrict__ reg,
                   const float* __restrict__ props,
                   float* __restrict__ scores_fg, float* __restrict__ boxes_fg,
                   const int* __restrict__ imh, const int* __restrict__ imw, int R) {
    int br = blockIdx.x;
    int b  = br / R, r = br % R;
    int c  = threadIdx.x;                        // 0..31 (classes 0..20 active)
    float s = (c < 21) ? logits[(size_t)br * 21 + c] : -3.0e38f;
    float mx = s;
    for (int o = 16; o > 0; o >>= 1) mx = fmaxf(mx, __shfl_xor(mx, o, 32));
    float e = (c < 21) ? expf(s - mx) : 0.0f;
    float sum = e;
    for (int o = 16; o > 0; o >>= 1) sum += __shfl_xor(sum, o, 32);
    if (c >= 1 && c <= 20) {
        const float* pr = props + (size_t)br * 4;
        float w  = pr[2] - pr[0], h = pr[3] - pr[1];
        float cx = pr[0] + 0.5f * w, cy = pr[1] + 0.5f * h;
        const float* d = reg + (size_t)br * 84 + c * 4;
        const float DWC = 4.135166556742356f;    // log(1000/16)
        float dx = d[0], dy = d[1];
        float dw = fminf(d[2], DWC), dh = fminf(d[3], DWC);
        float pcx = dx * w + cx, pcy = dy * h + cy;
        float pw  = expf(dw) * w, ph = expf(dh) * h;
        float Wi = (float)imw[0], Hi = (float)imh[0];
        float bx0 = fminf(fmaxf(pcx - 0.5f * pw, 0.0f), Wi);
        float by0 = fminf(fmaxf(pcy - 0.5f * ph, 0.0f), Hi);
        float bx1 = fminf(fmaxf(pcx + 0.5f * pw, 0.0f), Wi);
        float by1 = fminf(fmaxf(pcy + 0.5f * ph, 0.0f), Hi);
        size_t o = ((size_t)b * 20 + (c - 1)) * R + r;
        scores_fg[o] = e / sum;
        float* bo = boxes_fg + o * 4;
        bo[0] = bx0; bo[1] = by0; bo[2] = bx1; bo[3] = by1;
    }
}

// ---------------------------------------------------------------------------
// 5) per-(image,class) NMS: bitonic sort desc (stable ties) + greedy suppress
// ---------------------------------------------------------------------------
__global__ __launch_bounds__(256)
void nms_kernel(const float* __restrict__ scores_fg, const float* __restrict__ boxes_fg,
                float* __restrict__ sns, float* __restrict__ bns, int R) {
    __shared__ float key[512];
    __shared__ int   sidx[512];
    __shared__ float bx[512][4];
    __shared__ float area[512];
    __shared__ int   keep[512];
    int bc = blockIdx.x;                          // b*20 + class
    const float* sc  = scores_fg + (size_t)bc * R;
    const float* bxg = boxes_fg  + (size_t)bc * R * 4;

    for (int i = threadIdx.x; i < R; i += 256) {
        float v = sc[i];
        key[i]  = (v > 0.05f) ? v : -1.0f;
        sidx[i] = i;
    }
    __syncthreads();
    // bitonic sort: descending by key, ties -> smaller original index first
    for (int k = 2; k <= R; k <<= 1) {
        for (int j = k >> 1; j > 0; j >>= 1) {
            for (int i = threadIdx.x; i < R; i += 256) {
                int ixj = i ^ j;
                if (ixj > i) {
                    bool up = ((i & k) == 0);
                    float ka = key[i], kb = key[ixj];
                    int   ia = sidx[i], ib = sidx[ixj];
                    bool before = (ka > kb) || (ka == kb && ia < ib);
                    if (up != before) { key[i] = kb; key[ixj] = ka; sidx[i] = ib; sidx[ixj] = ia; }
                }
            }
            __syncthreads();
        }
    }
    for (int i = threadIdx.x; i < R; i += 256) {
        int o = sidx[i];
        float4 b4 = *(const float4*)(bxg + (size_t)o * 4);
        bx[i][0] = b4.x; bx[i][1] = b4.y; bx[i][2] = b4.z; bx[i][3] = b4.w;
        area[i] = (b4.z - b4.x) * (b4.w - b4.y);
        keep[i] = (key[i] > 0.05f) ? 1 : 0;
    }
    __syncthreads();
    for (int i = 0; i < R; ++i) {
        if (keep[i]) {
            float x1 = bx[i][0], y1 = bx[i][1], x2 = bx[i][2], y2 = bx[i][3];
            float ai = area[i];
            for (int jj = i + 1 + threadIdx.x; jj < R; jj += 256) {
                float xl = fmaxf(x1, bx[jj][0]), yt = fmaxf(y1, bx[jj][1]);
                float xr = fminf(x2, bx[jj][2]), yb = fminf(y2, bx[jj][3]);
                float inter = fmaxf(xr - xl, 0.0f) * fmaxf(yb - yt, 0.0f);
                float iou = inter / (ai + area[jj] - inter);
                if (iou > 0.5f) keep[jj] = 0;
            }
        }
        __syncthreads();
    }
    float* so = sns + (size_t)bc * R;
    float* bo = bns + (size_t)bc * R * 4;
    for (int i = threadIdx.x; i < R; i += 256) {
        so[i] = keep[i] ? key[i] : -1.0f;
        bo[i * 4 + 0] = bx[i][0]; bo[i * 4 + 1] = bx[i][1];
        bo[i * 4 + 2] = bx[i][2]; bo[i * 4 + 3] = bx[i][3];
    }
}

// ---------------------------------------------------------------------------
// 6) top-100 per image; out = boxes(B,100,4) | scores(B,100) | labels(B,100)
// ---------------------------------------------------------------------------
__global__ __launch_bounds__(256)
void topk_kernel(const float* __restrict__ sns, const float* __restrict__ bns,
                 float* __restrict__ out, int R) {
    __shared__ float vals[20 * 512];
    __shared__ float rv[256];
    __shared__ int   ri[256];
    int b = blockIdx.x;
    int N = 20 * R;
    const float* s = sns + (size_t)b * N;
    for (int i = threadIdx.x; i < N; i += 256) vals[i] = s[i];
    __syncthreads();
    int B = gridDim.x;
    float* obox = out + (size_t)b * 400;
    float* osc  = out + (size_t)B * 400 + (size_t)b * 100;
    int*   olab = (int*)out + (size_t)B * 500 + (size_t)b * 100;
    for (int t = 0; t < 100; ++t) {
        float bv = -3.4e38f;
        int   bi = N;
        for (int i = threadIdx.x; i < N; i += 256) {
            float v = vals[i];
            if (v > bv || (v == bv && i < bi)) { bv = v; bi = i; }
        }
        rv[threadIdx.x] = bv; ri[threadIdx.x] = bi;
        __syncthreads();
        for (int o = 128; o > 0; o >>= 1) {
            if (threadIdx.x < o) {
                float v2 = rv[threadIdx.x + o];
                int   i2 = ri[threadIdx.x + o];
                if (v2 > rv[threadIdx.x] || (v2 == rv[threadIdx.x] && i2 < ri[threadIdx.x])) {
                    rv[threadIdx.x] = v2; ri[threadIdx.x] = i2;
                }
            }
            __syncthreads();
        }
        if (threadIdx.x == 0) {
            float v = rv[0];
            int   i = ri[0];
            bool mask = v > -0.5f;
            const float* b4 = bns + ((size_t)b * N + i) * 4;
            obox[t * 4 + 0] = mask ? b4[0] : 0.0f;
            obox[t * 4 + 1] = mask ? b4[1] : 0.0f;
            obox[t * 4 + 2] = mask ? b4[2] : 0.0f;
            obox[t * 4 + 3] = mask ? b4[3] : 0.0f;
            osc[t]  = mask ? v : 0.0f;
            olab[t] = mask ? (i / R + 1) : 0;
            vals[i] = -3.4e38f;
        }
        __syncthreads();
    }
}

// ---------------------------------------------------------------------------
// launch
// ---------------------------------------------------------------------------
extern "C" void kernel_launch(void* const* d_in, const int* in_sizes, int n_in,
                              void* d_out, int out_size, void* d_ws, size_t ws_size,
                              hipStream_t stream) {
    const float* feat  = (const float*)d_in[0];
    const float* props = (const float*)d_in[1];
    const float* W6    = (const float*)d_in[2];
    const float* b6    = (const float*)d_in[3];
    const float* W7    = (const float*)d_in[4];
    const float* b7    = (const float*)d_in[5];
    const float* Wc    = (const float*)d_in[6];
    const float* bc    = (const float*)d_in[7];
    const float* Wb    = (const float*)d_in[8];
    const float* bb    = (const float*)d_in[9];
    const int*   imh   = (const int*)d_in[10];
    const int*   imw   = (const int*)d_in[11];

    const int B = 2, R = 512, D = 1024, IN = 12544;
    const int M = B * R;

    char* w = (char*)d_ws;
    auto carve = [&](size_t bytes) {
        char* p = w;
        w += (bytes + 255) & ~(size_t)255;
        return p;
    };
    float*          featT     = (float*)         carve((size_t)B * 32 * 32 * 256 * 4);
    unsigned short* X         = (unsigned short*)carve((size_t)M * IN * 2);
    unsigned short* Y1        = (unsigned short*)carve((size_t)M * D * 2);
    unsigned short* Y2        = (unsigned short*)carve((size_t)M * D * 2);
    float*          logits    = (float*)         carve((size_t)M * 21 * 4);
    float*          reg       = (float*)         carve((size_t)M * 84 * 4);
    float*          scores_fg = (float*)         carve((size_t)B * 20 * R * 4);
    float*          boxes_fg  = (float*)         carve((size_t)B * 20 * R * 4 * 4);
    float*          sns       = (float*)         carve((size_t)B * 20 * R * 4);
    float*          bns       = (float*)         carve((size_t)B * 20 * R * 4 * 4);

    transpose_feat_kernel<<<(B * 256 * 1024 + 255) / 256, 256, 0, stream>>>(feat, featT, B);
    roi_pool_kernel<<<B * R, 256, 0, stream>>>(featT, props, X, R);

    dim3 g1(D / BN, M / BM);
    gemm_bf16_nt_kernel<<<g1, 128, 0, stream>>>(X,  W6, b6, Y1, M, D, IN, 1);
    gemm_bf16_nt_kernel<<<g1, 128, 0, stream>>>(Y1, W7, b7, Y2, M, D, D, 1);

    head_kernel<<<(M * 105 + 255) / 256, 256, 0, stream>>>(Y2, Wc, bc, Wb, bb, logits, reg, M);
    decode_kernel<<<M, 32, 0, stream>>>(logits, reg, props, scores_fg, boxes_fg, imh, imw, R);
    nms_kernel<<<B * 20, 256, 0, stream>>>(scores_fg, boxes_fg, sns, bns, R);
    topk_kernel<<<B, 256, 0, stream>>>(sns, bns, (float*)d_out, R);
}
